// KANGNN_15496242004239
// MI455X (gfx1250) — matrix operand order repaired and checked by reading, not compiled, not run on hardware
//
#include <hip/hip_runtime.h>
#include <hip/hip_bf16.h>

#define N_NODES 100000
#define N_EDGES 1600000
#define DIM 64
#define NCOEF 8
#define KTOT 576            // 64 (silu) + 64*8 (spline bases)
#define NKT 18              // KTOT / 32
#define NLAYERS 6
#define ROWS_PER_WG 32
#define LDS_STRIDE 584      // 576 + 8 pad (ushorts)

typedef __bf16 bf16_t;
typedef bf16_t v16bf __attribute__((ext_vector_type(16)));
typedef float  v8f   __attribute__((ext_vector_type(8)));

__device__ __forceinline__ unsigned short f2bf(float f) {
    union { float f; unsigned u; } v; v.f = f;
    unsigned u = v.u;
    unsigned r = u + 0x7FFFu + ((u >> 16) & 1u);   // round-to-nearest-even
    return (unsigned short)(r >> 16);
}

// ---------------------------------------------------------------------------
// CSR row pointers from sorted COO rows (lower_bound per node)
// ---------------------------------------------------------------------------
__global__ __launch_bounds__(256) void build_rowptr(const int* __restrict__ er,
                                                    int* __restrict__ rp) {
    int i = blockIdx.x * blockDim.x + threadIdx.x;
    if (i > N_NODES) return;
    int lo = 0, hi = N_EDGES;
    while (lo < hi) {
        int mid = (lo + hi) >> 1;
        if (er[mid] < i) lo = mid + 1; else hi = mid;
    }
    rp[i] = lo;
}

// ---------------------------------------------------------------------------
// Pack combined weights W[k,o] (k<64: base_w[o][k]; else sw[o][d][q]*ss[o][d])
// into WMMA B-fragment order, bf16:
//   idx = (((layer*NKT + kt)*4 + nt)*32 + lane)*16 + (k&15)
//   lane = ((k&31)>=16 ? 16 : 0) + (o&15), per-lane 16 consecutive K values.
// ---------------------------------------------------------------------------
__global__ __launch_bounds__(256) void pack_w(const float* __restrict__ bw,
                                              const float* __restrict__ sw,
                                              const float* __restrict__ ss,
                                              unsigned short* __restrict__ wpk) {
    int id = blockIdx.x * blockDim.x + threadIdx.x;
    if (id >= NLAYERS * KTOT * DIM) return;
    int layer = id / (KTOT * DIM);
    int rem   = id - layer * (KTOT * DIM);
    int k = rem >> 6;          // 0..575
    int o = rem & 63;
    float w;
    if (k < DIM) {
        w = bw[(layer * DIM + o) * DIM + k];
    } else {
        int d = (k - DIM) >> 3, q = (k - DIM) & 7;
        w = sw[(((layer * DIM + o) * DIM) + d) * NCOEF + q]
          * ss[(layer * DIM + o) * DIM + d];
    }
    int kt = k >> 5, kr = k & 31;
    int nt = o >> 4, n = o & 15;
    int lane = ((kr >> 4) << 4) + n;
    int slot = kr & 15;
    wpk[(((layer * NKT + kt) * 4 + nt) * 32 + lane) * 16 + slot] = f2bf(w);
}

// ---------------------------------------------------------------------------
// SPMM: one wave32 per node, lane handles 2 dims, sequential (deterministic)
// x fits entirely in the 192MB L2, so the gather is L2-resident.
// ---------------------------------------------------------------------------
__global__ __launch_bounds__(256) void spmm_kernel(const int* __restrict__ rp,
                                                   const int* __restrict__ ecol,
                                                   const float* __restrict__ eval,
                                                   const float* __restrict__ xin,
                                                   float* __restrict__ yout) {
    int wave = threadIdx.x >> 5, lane = threadIdx.x & 31;
    int node = blockIdx.x * 8 + wave;
    if (node >= N_NODES) return;
    int s = rp[node], e = rp[node + 1];
    float ax = 0.0f, ay = 0.0f;
    for (int i = s; i < e; ++i) {
        float v = eval[i];
        int   c = ecol[i];
        const float2 xv = *(const float2*)(xin + c * DIM + lane * 2);
        ax = fmaf(v, xv.x, ax);
        ay = fmaf(v, xv.y, ay);
    }
    float2 r; r.x = ax; r.y = ay;
    *(float2*)(yout + node * DIM + lane * 2) = r;
}

// ---------------------------------------------------------------------------
// Fused KANLinear:
//  Phase 1: build [32 x 576] bf16 activation tile (silu + B-spline bases) in LDS
//  Phase 2: 8 waves x one 16x16 output tile each, 18 bf16 WMMA steps (K=576)
// ---------------------------------------------------------------------------
__global__ __launch_bounds__(256) void kan_kernel(const float* __restrict__ xin,
                                                  const unsigned short* __restrict__ wpk,
                                                  float* __restrict__ out) {
    __shared__ unsigned short fLDS[ROWS_PER_WG * LDS_STRIDE];
    const int tid   = threadIdx.x;
    const int wrow0 = blockIdx.x * ROWS_PER_WG;

    // ---- Phase 1: activations -> LDS (bf16) ----
    {
        int r    = tid >> 3;            // 0..31
        int dcol = (tid & 7) << 3;      // 0,8,...,56
        int rg   = wrow0 + r;
        float xv[8];
        if (rg < N_NODES) {
            const float4* xp = (const float4*)(xin + rg * DIM + dcol);
            float4 a = xp[0], b = xp[1];
            xv[0]=a.x; xv[1]=a.y; xv[2]=a.z; xv[3]=a.w;
            xv[4]=b.x; xv[5]=b.y; xv[6]=b.z; xv[7]=b.w;
        } else {
            #pragma unroll
            for (int j = 0; j < 8; ++j) xv[j] = 0.0f;
        }
        unsigned short* frow = &fLDS[r * LDS_STRIDE];
        const float h = 0.4f;
        #pragma unroll
        for (int j = 0; j < 8; ++j) {
            float xx = xv[j];
            // SiLU branch
            frow[dcol + j] = f2bf(xx / (1.0f + __expf(-xx)));
            // Cox-de Boor, uniform grid t[i] = (i-3)*h - 1, i=0..11
            float b[11];
            #pragma unroll
            for (int q = 0; q < 11; ++q) {
                float t0 = (float)(q - 3) * h - 1.0f;
                b[q] = (xx >= t0 && xx < t0 + h) ? 1.0f : 0.0f;
            }
            #pragma unroll
            for (int p = 1; p <= 3; ++p) {
                float inv = 1.0f / ((float)p * h);
                #pragma unroll
                for (int q = 0; q + p < 11; ++q) {
                    float tq   = (float)(q - 3) * h - 1.0f;
                    float tqp1 = (float)(q + p + 1 - 3) * h - 1.0f;
                    b[q] = (xx - tq) * inv * b[q] + (tqp1 - xx) * inv * b[q + 1];
                }
            }
            #pragma unroll
            for (int k = 0; k < NCOEF; ++k)
                frow[DIM + (dcol + j) * NCOEF + k] = f2bf(b[k]);
        }
    }
    __syncthreads();

    // ---- Phase 2: WMMA GEMM, one 16x16 tile per wave ----
    const int wave = tid >> 5, lane = tid & 31;
    const int rowb = (wave & 1) << 4;     // 0 or 16
    const int colb = (wave >> 1) << 4;    // 0,16,32,48
    const int m     = lane & 15;
    const int khalf = (lane >> 4) << 3;   // 0 or 8 (A-fragment half select)

    v8f acc = {0.f, 0.f, 0.f, 0.f, 0.f, 0.f, 0.f, 0.f};
    const unsigned short* frow = &fLDS[(rowb + m) * LDS_STRIDE];
    const int nt = colb >> 4;

    for (int kt = 0; kt < NKT; ++kt) {
        // A fragment: lane m holds K = {2*khalf..2*khalf+15} (contiguous, matches packed B)
        union { v16bf v; unsigned u[8]; } A;
        #pragma unroll
        for (int j = 0; j < 8; ++j) {
            int kk = kt * 32 + (khalf << 1) + (j << 1);
            A.u[j] = *(const unsigned*)(frow + kk);
        }
        // B fragment: pre-swizzled, 32 contiguous bytes per lane
        union { v16bf v; uint4 q[2]; } B;
        const uint4* bp = (const uint4*)(wpk + (((kt << 2) + nt) * 32 + lane) * 16);
        B.q[0] = bp[0];
        B.q[1] = bp[1];
        acc = __builtin_amdgcn_wmma_f32_16x16x32_bf16(
                  false, A.v, false, B.v, (short)0, acc, false, false);
    }

    // D layout: VGPR j -> row (j or 8+j), col = lane&15 (per 16x16 f32 C/D map)
    const int r0 = wrow0 + rowb + ((lane >> 4) << 3);
    const int c  = colb + m;
    #pragma unroll
    for (int j = 0; j < 8; ++j) {
        int rr = r0 + j;
        if (rr < N_NODES) out[rr * DIM + c] = acc[j];
    }
}

// ---------------------------------------------------------------------------
// Host orchestration (graph-capture safe: only kernel launches on stream)
// ---------------------------------------------------------------------------
extern "C" void kernel_launch(void* const* d_in, const int* in_sizes, int n_in,
                              void* d_out, int out_size, void* d_ws, size_t ws_size,
                              hipStream_t stream) {
    const float* x        = (const float*)d_in[0];
    const float* edge_val = (const float*)d_in[1];
    const float* base_w   = (const float*)d_in[2];
    const float* spline_w = (const float*)d_in[3];
    const float* spline_s = (const float*)d_in[4];
    const int*   edge_row = (const int*)d_in[5];
    const int*   edge_col = (const int*)d_in[6];
    float* out = (float*)d_out;

    const size_t NODE_BYTES = (size_t)N_NODES * DIM * sizeof(float); // 25.6 MB
    char* ws = (char*)d_ws;
    float* bufA = (float*)ws;
    float* bufB = (float*)(ws + NODE_BYTES);
    int*   rowptr = (int*)(ws + 2 * NODE_BYTES);
    unsigned short* wpk = (unsigned short*)(ws + 2 * NODE_BYTES + 400128 /* rowptr, 128B-aligned */);

    build_rowptr<<<(N_NODES + 1 + 255) / 256, 256, 0, stream>>>(edge_row, rowptr);
    pack_w<<<(NLAYERS * KTOT * DIM + 255) / 256, 256, 0, stream>>>(base_w, spline_w, spline_s, wpk);

    const int spmm_grid = (N_NODES + 7) / 8;                 // 12500
    const int kan_grid  = (N_NODES + ROWS_PER_WG - 1) / ROWS_PER_WG; // 3125
    const size_t WPK_LAYER = (size_t)NKT * 4 * 32 * 16;      // ushorts per layer

    const float* cur = x;
    float* bufs[2] = {bufA, bufB};
    int pp = 0, layer = 0;
    for (int blk = 0; blk < 3; ++blk) {
        float* y = bufs[pp]; pp ^= 1;
        spmm_kernel<<<spmm_grid, 256, 0, stream>>>(rowptr, edge_col, edge_val, cur, y);
        cur = y;
        for (int l = 0; l < 2; ++l) {
            float* o = (blk == 2 && l == 1) ? out : bufs[pp];
            if (!(blk == 2 && l == 1)) pp ^= 1;
            kan_kernel<<<kan_grid, 256, 0, stream>>>(cur, wpk + (size_t)layer * WPK_LAYER, o);
            cur = o; ++layer;
        }
    }
}
